// MultiHeadAttention_70635032150575
// MI455X (gfx1250) — compile-verified, matrix operand order checked by reading, not run on hardware
//
#include <hip/hip_runtime.h>
#include <hip/hip_bf16.h>

// ---------------------------------------------------------------------------
// MI455X (gfx1250) fused multi-head attention:
//   proj_kernel  x3 : Q/K/V projections fp32 -> bf16 head layout (V transposed)
//   attn_kernel     : flash attention, v_wmma_f32_16x16x32_bf16 for QK^T / PV,
//                     K/V tiles double-buffered via global_load_async_to_lds
//   oproj_kernel    : output projection bf16 -> fp32 + bias (async A staging)
// ---------------------------------------------------------------------------

typedef __attribute__((ext_vector_type(16))) __bf16 v16bf;
typedef __attribute__((ext_vector_type(8)))  __bf16 v8bf;
typedef __attribute__((ext_vector_type(8)))  float  v8f;

#define D_MODEL 768
#define SEQ     2048
#define NHEAD   12
#define DK      64

#define P_TM  128
#define P_TN  64
#define P_LDA 40      // 32 + 8 pad (bf16 elems), 80B rows -> 16B aligned

__device__ __forceinline__ v8f vzero8() {
  v8f z = {0.f,0.f,0.f,0.f,0.f,0.f,0.f,0.f};
  return z;
}

// Async 16-byte global -> LDS copy (ASYNCcnt-tracked, bypasses VGPRs).
// dst = LDS byte offset (low 32 bits of generic pointer), src = global addr.
__device__ __forceinline__ void async_copy_b128(unsigned dst_lds, const void* src) {
  asm volatile("global_load_async_to_lds_b128 %0, %1, off"
               :: "v"(dst_lds), "v"(src) : "memory");
}
__device__ __forceinline__ unsigned lds_addr(const void* p) {
  return (unsigned)(unsigned long long)p;
}

// Load one 16x32 bf16 operand fragment (A pattern; B^T uses the same pattern).
// Lane l: row = l&15, K-half = (l>>4)*8; elems [0..7] = K(kh..kh+7),
// [8..15] = K(16+kh..16+kh+7).  base/stride must keep 16B alignment.
__device__ __forceinline__ v16bf load_frag(const __bf16* base, int stride, int lane) {
  const int row = lane & 15;
  const int kh  = (lane >> 4) << 3;
  const __bf16* p = base + row * stride;
  v8bf lo = *(const v8bf*)(p + kh);
  v8bf hi = *(const v8bf*)(p + 16 + kh);
  v16bf r;
#pragma unroll
  for (int i = 0; i < 8; ++i) { r[i] = lo[i]; r[i + 8] = hi[i]; }
  return r;
}

__device__ __forceinline__ v8f wmma_bf16(v16bf a, v16bf b, v8f c) {
  return __builtin_amdgcn_wmma_f32_16x16x32_bf16(false, a, false, b,
                                                 (short)0, c, false, false);
}

// ---------------------------------------------------------------------------
// Projection: out = X @ W^T + bias, X:[B*S,768] fp32, W:[768,768] fp32.
// mode 0: Q  -> [B,H,S,64] bf16, scaled by 1/sqrt(64)
// mode 1: K  -> [B,H,S,64] bf16
// mode 2: V  -> [B,H,64,S] bf16 (transposed for PV WMMA B-operand)
// ---------------------------------------------------------------------------
__global__ __launch_bounds__(256) void proj_kernel(
    const float* __restrict__ X, const float* __restrict__ W,
    const float* __restrict__ bias, __bf16* __restrict__ out, int mode)
{
  __shared__ __bf16 Xl[P_TM * P_LDA];
  __shared__ __bf16 Wl[P_TN * P_LDA];

  const int tid  = threadIdx.x;
  const int lane = tid & 31;
  const int wave = tid >> 5;        // 0..7
  const int wm   = wave & 3;        // 4 waves along M
  const int wn   = wave >> 2;       // 2 waves along N
  const int m0   = blockIdx.y * P_TM;
  const int n0   = blockIdx.x * P_TN;

  const int srow = tid >> 3;            // staging row (0..31), +32 per chunk
  const int soff = (tid & 7) << 2;      // staging col offset (floats)

  v8f acc[2][2];
#pragma unroll
  for (int i = 0; i < 2; ++i)
#pragma unroll
    for (int j = 0; j < 2; ++j) acc[i][j] = vzero8();

  for (int k0 = 0; k0 < D_MODEL; k0 += 32) {
    __syncthreads();
    // Batched static-trip staging: issue all global loads, then convert+store.
    float4 fx[4];
#pragma unroll
    for (int it = 0; it < 4; ++it)
      fx[it] = *(const float4*)(X + (size_t)(m0 + srow + it * 32) * D_MODEL + k0 + soff);
    float4 fw[2];
#pragma unroll
    for (int it = 0; it < 2; ++it)
      fw[it] = *(const float4*)(W + (size_t)(n0 + srow + it * 32) * D_MODEL + k0 + soff);
#pragma unroll
    for (int it = 0; it < 4; ++it) {
      __bf16* dp = Xl + (srow + it * 32) * P_LDA + soff;
      dp[0] = (__bf16)fx[it].x; dp[1] = (__bf16)fx[it].y;
      dp[2] = (__bf16)fx[it].z; dp[3] = (__bf16)fx[it].w;
    }
#pragma unroll
    for (int it = 0; it < 2; ++it) {
      __bf16* dp = Wl + (srow + it * 32) * P_LDA + soff;
      dp[0] = (__bf16)fw[it].x; dp[1] = (__bf16)fw[it].y;
      dp[2] = (__bf16)fw[it].z; dp[3] = (__bf16)fw[it].w;
    }
    __syncthreads();

    v16bf a0 = load_frag(Xl + (wm * 32     ) * P_LDA, P_LDA, lane);
    v16bf a1 = load_frag(Xl + (wm * 32 + 16) * P_LDA, P_LDA, lane);
    v16bf b0 = load_frag(Wl + (wn * 32     ) * P_LDA, P_LDA, lane);
    v16bf b1 = load_frag(Wl + (wn * 32 + 16) * P_LDA, P_LDA, lane);
    acc[0][0] = wmma_bf16(a0, b0, acc[0][0]);
    acc[0][1] = wmma_bf16(a0, b1, acc[0][1]);
    acc[1][0] = wmma_bf16(a1, b0, acc[1][0]);
    acc[1][1] = wmma_bf16(a1, b1, acc[1][1]);
  }

  const int hl = lane >> 4;
  const int cn = lane & 15;
  const float qs = (mode == 0) ? 0.125f : 1.0f;   // 1/sqrt(64) folded into Q
#pragma unroll
  for (int i = 0; i < 2; ++i)
#pragma unroll
    for (int j = 0; j < 2; ++j) {
      const int n  = n0 + wn * 32 + j * 16 + cn;
      const float bb = bias[n];
      const int h = n >> 6, d = n & 63;
#pragma unroll
      for (int r = 0; r < 8; ++r) {
        const int m = m0 + wm * 32 + i * 16 + r + 8 * hl;
        const int b = m >> 11;           // m / 2048
        const int s = m & 2047;
        const float val = (acc[i][j][r] + bb) * qs;
        const size_t idx = (mode == 2)
            ? (((size_t)(b * NHEAD + h) * DK + d) * SEQ + s)     // V^T layout
            : (((size_t)(b * NHEAD + h) * SEQ + s) * DK + d);    // head layout
        out[idx] = (__bf16)val;
      }
    }
}

// ---------------------------------------------------------------------------
// Flash attention: one (b,h) x 64-query-row tile per 128-thread block.
// Wave w owns query rows [qt*64 + w*16, +16). Online softmax, fp32 state.
// K/V tiles double-buffered through the async global->LDS path.
// ---------------------------------------------------------------------------
#define A_LD  72     // 64 + 8 pad (bf16 elems), 144B rows -> 16B aligned
#define NKV   (SEQ / 64)

__device__ __forceinline__ void stage_kv_async(
    const __bf16* __restrict__ Kbase, const __bf16* __restrict__ Vbase,
    int kv0, __bf16* Kd, __bf16* Vd, int tid)
{
  const int row0 = tid >> 3;           // 0..15, +16 per chunk
  const int off  = (tid & 7) << 3;     // bf16 elems, 16B aligned
#pragma unroll
  for (int it = 0; it < 4; ++it) {
    const int row = row0 + it * 16;
    async_copy_b128(lds_addr(Kd + row * A_LD + off),
                    Kbase + (size_t)(kv0 + row) * DK + off);
    async_copy_b128(lds_addr(Vd + row * A_LD + off),
                    Vbase + (size_t)row * SEQ + kv0 + off);
  }
}

__global__ __launch_bounds__(128) void attn_kernel(
    const __bf16* __restrict__ Qh, const __bf16* __restrict__ Kh,
    const __bf16* __restrict__ Vt, __bf16* __restrict__ Ccat)
{
  __shared__ __bf16 Kl[2][64 * A_LD];     // K tiles: 64 kv rows x 64 d
  __shared__ __bf16 Vl[2][64 * A_LD];     // V^T tiles: 64 d rows x 64 kv
  __shared__ __bf16 Pl[4 * 16 * A_LD];    // per-wave P transpose buffer

  const int tid  = threadIdx.x;
  const int lane = tid & 31;
  const int wave = tid >> 5;              // 0..3
  const int qt   = blockIdx.x;            // query tile
  const int bh   = blockIdx.y;
  const int b    = bh / NHEAD;
  const int h    = bh % NHEAD;

  const __bf16* Qbase = Qh + (size_t)bh * SEQ * DK;
  const __bf16* Kbase = Kh + (size_t)bh * SEQ * DK;
  const __bf16* Vbase = Vt + (size_t)bh * DK * SEQ;
  __bf16* Pw = Pl + wave * 16 * A_LD;

  // Q fragments for this wave's 16 rows (2 k-steps of 32), kept in VGPRs.
  v16bf qf[2];
  {
    const int row = qt * 64 + wave * 16 + (lane & 15);
    const int kh  = (lane >> 4) << 3;
    const __bf16* p = Qbase + (size_t)row * DK;
#pragma unroll
    for (int ks = 0; ks < 2; ++ks) {
      v8bf lo = *(const v8bf*)(p + ks * 32 + kh);
      v8bf hi = *(const v8bf*)(p + ks * 32 + 16 + kh);
#pragma unroll
      for (int i = 0; i < 8; ++i) { qf[ks][i] = lo[i]; qf[ks][i + 8] = hi[i]; }
    }
  }

  const int hl = lane >> 4;
  const int cn = lane & 15;
  float mrow[8], lrow[8];
  v8f o[4];
#pragma unroll
  for (int r = 0; r < 8; ++r) { mrow[r] = -1e30f; lrow[r] = 0.f; }
#pragma unroll
  for (int t = 0; t < 4; ++t) o[t] = vzero8();

  // Prefetch tile 0 into buffer 0.
  stage_kv_async(Kbase, Vbase, 0, Kl[0], Vl[0], tid);

  for (int i = 0; i < NKV; ++i) {
    const int cur = i & 1;
    // Issue next tile's DMA into the other buffer (safe: bottom barrier of the
    // previous iteration guarantees nobody still reads it), then retire the
    // oldest 8 transfers (= this tile's copies). ASYNCcnt completes in order.
    if (i + 1 < NKV) {
      stage_kv_async(Kbase, Vbase, (i + 1) * 64, Kl[1 - cur], Vl[1 - cur], tid);
      asm volatile("s_wait_asynccnt 0x8" ::: "memory");
    } else {
      asm volatile("s_wait_asynccnt 0x0" ::: "memory");
    }
    __syncthreads();
    const __bf16* Kc = Kl[cur];
    const __bf16* Vc = Vl[cur];

    // Scores S = (Q*scale) K^T : 16 x 64, 4 n-tiles, 2 k-steps each.
    v8f sc[4];
#pragma unroll
    for (int t = 0; t < 4; ++t) sc[t] = vzero8();
#pragma unroll
    for (int ks = 0; ks < 2; ++ks)
#pragma unroll
      for (int t = 0; t < 4; ++t) {
        v16bf bf = load_frag(Kc + (t * 16) * A_LD + ks * 32, A_LD, lane);
        sc[t] = wmma_bf16(qf[ks], bf, sc[t]);
      }

    // Online softmax. Lane holds row (r + 8*hl), column (t*16 + cn);
    // full row lives in the 16 lanes of this half -> xor-reduce masks 1,2,4,8.
    float alpha[8];
#pragma unroll
    for (int r = 0; r < 8; ++r) {
      float mx = fmaxf(fmaxf(sc[0][r], sc[1][r]), fmaxf(sc[2][r], sc[3][r]));
      mx = fmaxf(mx, __shfl_xor(mx, 1));
      mx = fmaxf(mx, __shfl_xor(mx, 2));
      mx = fmaxf(mx, __shfl_xor(mx, 4));
      mx = fmaxf(mx, __shfl_xor(mx, 8));
      const float mn = fmaxf(mrow[r], mx);
      alpha[r] = __expf(mrow[r] - mn);
      mrow[r]  = mn;
      float rs = 0.f;
#pragma unroll
      for (int t = 0; t < 4; ++t) {
        const float p = __expf(sc[t][r] - mn);
        sc[t][r] = p;
        rs += p;
      }
      rs += __shfl_xor(rs, 1);
      rs += __shfl_xor(rs, 2);
      rs += __shfl_xor(rs, 4);
      rs += __shfl_xor(rs, 8);
      lrow[r] = lrow[r] * alpha[r] + rs;
    }

    // Rescale O; dump P (C-layout) to per-wave LDS to re-read in A-layout.
#pragma unroll
    for (int t = 0; t < 4; ++t)
#pragma unroll
      for (int r = 0; r < 8; ++r) {
        o[t][r] *= alpha[r];
        Pw[(r + 8 * hl) * A_LD + t * 16 + cn] = (__bf16)sc[t][r];
      }
    // Same-wave LDS RAW is processed in order on CDNA5; no barrier needed.

    // O += P V : A = P (16 x 64 kv), B^T = V^T rows (d x kv).
#pragma unroll
    for (int ks = 0; ks < 2; ++ks) {
      v16bf pa = load_frag(Pw + ks * 32, A_LD, lane);
#pragma unroll
      for (int t = 0; t < 4; ++t) {
        v16bf vb = load_frag(Vc + (t * 16) * A_LD + ks * 32, A_LD, lane);
        o[t] = wmma_bf16(pa, vb, o[t]);
      }
    }
    __syncthreads();   // all reads of buffer `cur` done before it is re-filled
  }

  // Normalize and write concat[b][s][h*64 + d] in bf16.
#pragma unroll
  for (int r = 0; r < 8; ++r) {
    const float inv = 1.0f / lrow[r];
    const int s = qt * 64 + wave * 16 + r + 8 * hl;
#pragma unroll
    for (int t = 0; t < 4; ++t) {
      const size_t idx =
          ((size_t)b * SEQ + s) * D_MODEL + h * DK + t * 16 + cn;
      Ccat[idx] = (__bf16)(o[t][r] * inv);
    }
  }
}

// ---------------------------------------------------------------------------
// Output projection: out = concat @ Wo^T + bo, concat bf16 (async staged),
// Wo fp32 -> bf16 in VGPRs, out fp32.
// ---------------------------------------------------------------------------
__global__ __launch_bounds__(256) void oproj_kernel(
    const __bf16* __restrict__ A, const float* __restrict__ W,
    const float* __restrict__ bias, float* __restrict__ out)
{
  __shared__ __bf16 Al[P_TM * P_LDA];
  __shared__ __bf16 Wl[P_TN * P_LDA];

  const int tid  = threadIdx.x;
  const int lane = tid & 31;
  const int wave = tid >> 5;
  const int wm   = wave & 3;
  const int wn   = wave >> 2;
  const int m0   = blockIdx.y * P_TM;
  const int n0   = blockIdx.x * P_TN;

  const int wrow = tid >> 3;            // W staging row, +32 per chunk
  const int woff = (tid & 7) << 2;
  const int arow = tid >> 2;            // A staging row, +64 per chunk
  const int aoff = (tid & 3) << 3;      // bf16 elems

  v8f acc[2][2];
#pragma unroll
  for (int i = 0; i < 2; ++i)
#pragma unroll
    for (int j = 0; j < 2; ++j) acc[i][j] = vzero8();

  for (int k0 = 0; k0 < D_MODEL; k0 += 32) {
    __syncthreads();
    // A tile: pure bf16 copy -> async DMA straight into LDS.
#pragma unroll
    for (int it = 0; it < 2; ++it) {
      const int row = arow + it * 64;
      async_copy_b128(lds_addr(Al + row * P_LDA + aoff),
                      A + (size_t)(m0 + row) * D_MODEL + k0 + aoff);
    }
    // W tile: fp32 -> bf16 through VGPRs (batched loads, then converts).
    float4 fw[2];
#pragma unroll
    for (int it = 0; it < 2; ++it)
      fw[it] = *(const float4*)(W + (size_t)(n0 + wrow + it * 32) * D_MODEL + k0 + woff);
#pragma unroll
    for (int it = 0; it < 2; ++it) {
      __bf16* dp = Wl + (wrow + it * 32) * P_LDA + woff;
      dp[0] = (__bf16)fw[it].x; dp[1] = (__bf16)fw[it].y;
      dp[2] = (__bf16)fw[it].z; dp[3] = (__bf16)fw[it].w;
    }
    asm volatile("s_wait_asynccnt 0x0" ::: "memory");
    __syncthreads();

    v16bf a0 = load_frag(Al + (wm * 32     ) * P_LDA, P_LDA, lane);
    v16bf a1 = load_frag(Al + (wm * 32 + 16) * P_LDA, P_LDA, lane);
    v16bf b0 = load_frag(Wl + (wn * 32     ) * P_LDA, P_LDA, lane);
    v16bf b1 = load_frag(Wl + (wn * 32 + 16) * P_LDA, P_LDA, lane);
    acc[0][0] = wmma_bf16(a0, b0, acc[0][0]);
    acc[0][1] = wmma_bf16(a0, b1, acc[0][1]);
    acc[1][0] = wmma_bf16(a1, b0, acc[1][0]);
    acc[1][1] = wmma_bf16(a1, b1, acc[1][1]);
  }

  const int hl = lane >> 4;
  const int cn = lane & 15;
#pragma unroll
  for (int i = 0; i < 2; ++i)
#pragma unroll
    for (int j = 0; j < 2; ++j) {
      const int n = n0 + wn * 32 + j * 16 + cn;
      const float bb = bias[n];
#pragma unroll
      for (int r = 0; r < 8; ++r) {
        const int m = m0 + wm * 32 + i * 16 + r + 8 * hl;
        out[(size_t)m * D_MODEL + n] = acc[i][j][r] + bb;
      }
    }
}

// ---------------------------------------------------------------------------
extern "C" void kernel_launch(void* const* d_in, const int* in_sizes, int n_in,
                              void* d_out, int out_size, void* d_ws, size_t ws_size,
                              hipStream_t stream)
{
  const float* q  = (const float*)d_in[0];
  const float* k  = (const float*)d_in[1];
  const float* v  = (const float*)d_in[2];
  const float* Wq = (const float*)d_in[3];
  const float* bq = (const float*)d_in[4];
  const float* Wk = (const float*)d_in[5];
  const float* bk = (const float*)d_in[6];
  const float* Wv = (const float*)d_in[7];
  const float* bv = (const float*)d_in[8];
  const float* Wo = (const float*)d_in[9];
  const float* bo = (const float*)d_in[10];
  float* out = (float*)d_out;

  const int B = in_sizes[0] / (SEQ * D_MODEL);   // = 2
  const size_t per = (size_t)B * NHEAD * SEQ * DK;
  __bf16* Qh = (__bf16*)d_ws;
  __bf16* Kh = Qh + per;
  __bf16* Vt = Kh + per;
  __bf16* Cc = Vt + per;                         // ~24 MB total workspace

  dim3 gproj(D_MODEL / P_TN, (B * SEQ) / P_TM);  // (12, 32)
  proj_kernel<<<gproj, 256, 0, stream>>>(q, Wq, bq, Qh, 0);
  proj_kernel<<<gproj, 256, 0, stream>>>(k, Wk, bk, Kh, 1);
  proj_kernel<<<gproj, 256, 0, stream>>>(v, Wv, bv, Vt, 2);

  dim3 gattn(SEQ / 64, B * NHEAD);               // (32, 24)
  attn_kernel<<<gattn, 128, 0, stream>>>(Qh, Kh, Vt, Cc);

  oproj_kernel<<<gproj, 256, 0, stream>>>(Cc, Wo, bo, out);
}
